// PairConditionedRetriever_21242908246103
// MI455X (gfx1250) — matrix operand order repaired and checked by reading, not compiled, not run on hardware
//
#include <hip/hip_runtime.h>
#include <math.h>

// ---------------------------------------------------------------------------
// PairConditionedRetriever for MI455X (gfx1250, wave32, WMMA + TDM)
// B=2048, K=64, E=512, S=16
// ---------------------------------------------------------------------------

#define BATCH 2048
#define KNBR  64
#define EDIM  512
#define SDIM  16
#define KEXT  544   // 512 (neighbors) + 16 (stats) + 16 zero pad -> multiple of 32
#define KQC   1024  // query|condition concat

typedef __attribute__((ext_vector_type(16))) __bf16 v16bf;
typedef __attribute__((ext_vector_type(8)))  __bf16 v8bf;
typedef __attribute__((ext_vector_type(8)))  float  v8f;
typedef __attribute__((ext_vector_type(4)))  unsigned int v4u;
typedef __attribute__((ext_vector_type(8)))  int v8i;
typedef __attribute__((ext_vector_type(4)))  int v4i;

#if defined(__has_builtin)
#if __has_builtin(__builtin_amdgcn_tensor_load_to_lds)
#define PCR_USE_TDM 1
#endif
#endif
#ifndef PCR_USE_TDM
#define PCR_USE_TDM 0
#endif

#define CAT16(lo, hi) __builtin_shufflevector(lo, hi, 0,1,2,3,4,5,6,7,8,9,10,11,12,13,14,15)

__device__ __forceinline__ float gelu_exact(float x) {
    return 0.5f * x * (1.0f + erff(x * 0.70710678118654752f));
}

#if PCR_USE_TDM
// Issue one TDM 2D tile load: 32 (K, 2B elems) x 512 (N rows) slice of the
// pre-transposed weight matrix Wt[512][KPAD] -> LDS laid out as [n][32].
// D# bit layout per CDNA5 ISA ch.8 (group0 128b, group1 256b, groups2/3 zero;
// 6-arg clang-23 builtin takes an extra zero int32x8 before the cache policy).
template<int KPAD>
__device__ __forceinline__ void tdm_stage_wslice(const __bf16* Wt, int ks,
                                                 unsigned lds_byte_addr) {
    unsigned long long ga =
        (unsigned long long)(uintptr_t)Wt + (unsigned long long)ks * 64ull; // ks*32 elems *2B
    v4u g0 = { 1u,                                   // count=1 (valid user D#)
               lds_byte_addr,                        // lds_addr
               (unsigned)ga,                         // global_addr[31:0]
               ((unsigned)(ga >> 32) & 0x01FFFFFFu)  // global_addr[56:32]
                   | (2u << 30) };                   // type=2 ("image")
    v8i g1 = { (int)(1u << 16),       // data_size=1 (2 bytes); wg_mask=0
               (int)(32u << 16),      // tensor_dim0 = 32  (bits 79:48)
               (int)(512u << 16),     // tensor_dim1 = 512 (bits 111:80)
               (int)(32u << 16),      // tile_dim0 = 32    (bits 127:112)
               512,                   // tile_dim1 = 512   (bits 143:128)
               (int)KPAD,             // tensor_dim0_stride (bits 207:160)
               0, 0 };                // dim1_stride unused (2D)
    v4i gz4 = { 0, 0, 0, 0 };
    v8i gz8 = { 0, 0, 0, 0, 0, 0, 0, 0 };
    __builtin_amdgcn_tensor_load_to_lds(g0, g1, gz4, gz4, gz8, 0);
}
#endif

// ---------------------------------------------------------------------------
// Core block GEMM: C[64, 512] += A[64, KPAD](bf16) @ Wt^T, Wt bf16 [512][KPAD].
// Block = 512 threads = 16 waves. Wave w: M-tile = w&3, N-tiles = (w>>2)*8..+8.
// Weight K-slices double-buffered in LDS; staged by TDM (wave 0) when present.
// ---------------------------------------------------------------------------
template<int KPAD, int KSTEPS>
__device__ __forceinline__ void gemm64x512(const __bf16* __restrict__ A,
                                           const __bf16* __restrict__ Wt,
                                           __bf16* Bs,   // LDS, 2 x 512*32 bf16
                                           v8f acc[8]) {
    const int tid  = threadIdx.x;
    const int lane = tid & 31;
    const int wave = tid >> 5;
    const int mt   = wave & 3;
    const int nb   = (wave >> 2) * 128;
    const int lr   = lane & 15;
    const int akh  = (lane >> 4) * 16;   // A: K-half offset (0 or 16)
    const int bkh  = (lane >> 4) * 16;   // B: K-half offset (0 or 16)
    const __bf16* Arow = A + (size_t)(mt * 16 + lr) * KPAD;

#if PCR_USE_TDM
    const unsigned lds0 = (unsigned)(uintptr_t)Bs;          // LDS byte offset
    if (wave == 0) {
        tdm_stage_wslice<KPAD>(Wt, 0, lds0);
        __builtin_amdgcn_s_wait_tensorcnt(0);
    }
    __syncthreads();
    for (int ks = 0; ks < KSTEPS; ++ks) {
        if (wave == 0 && ks + 1 < KSTEPS)                    // prefetch next slice
            tdm_stage_wslice<KPAD>(Wt, ks + 1, lds0 + (unsigned)(((ks + 1) & 1) * (EDIM * 32 * 2)));
        const __bf16* Bcur = Bs + (size_t)(ks & 1) * (EDIM * 32);
        const int k0 = ks * 32;
        v8bf alo = *(const v8bf*)(Arow + k0 + akh);
        v8bf ahi = *(const v8bf*)(Arow + k0 + akh + 8);
        v16bf av = CAT16(alo, ahi);
        v16bf bfr[8];
#pragma unroll
        for (int nt = 0; nt < 8; ++nt) {
            const __bf16* Br = Bcur + (size_t)(nb + nt * 16 + lr) * 32;
            v8bf blo = *(const v8bf*)(Br + bkh);
            v8bf bhi = *(const v8bf*)(Br + bkh + 8);
            bfr[nt] = CAT16(blo, bhi);
        }
#pragma unroll
        for (int nt = 0; nt < 8; ++nt)
            acc[nt] = __builtin_amdgcn_wmma_f32_16x16x32_bf16(
                false, av, false, bfr[nt], (short)0, acc[nt], false, false);
        if (wave == 0 && ks + 1 < KSTEPS)
            __builtin_amdgcn_s_wait_tensorcnt(0);
        __syncthreads();
    }
#else
    for (int ks = 0; ks < KSTEPS; ++ks) {
        __syncthreads();
        {   // manual staging fallback
            const v8bf* src = (const v8bf*)(Wt + (size_t)tid * KPAD + ks * 32);
            v8bf* dst = (v8bf*)(Bs + tid * 32);
            dst[0] = src[0]; dst[1] = src[1]; dst[2] = src[2]; dst[3] = src[3];
        }
        __syncthreads();
        const int k0 = ks * 32;
        v8bf alo = *(const v8bf*)(Arow + k0 + akh);
        v8bf ahi = *(const v8bf*)(Arow + k0 + akh + 8);
        v16bf av = CAT16(alo, ahi);
        v16bf bfr[8];
#pragma unroll
        for (int nt = 0; nt < 8; ++nt) {
            const __bf16* Br = Bs + (size_t)(nb + nt * 16 + lr) * 32;
            v8bf blo = *(const v8bf*)(Br + bkh);
            v8bf bhi = *(const v8bf*)(Br + bkh + 8);
            bfr[nt] = CAT16(blo, bhi);
        }
#pragma unroll
        for (int nt = 0; nt < 8; ++nt)
            acc[nt] = __builtin_amdgcn_wmma_f32_16x16x32_bf16(
                false, av, false, bfr[nt], (short)0, acc[nt], false, false);
    }
    __syncthreads();
#endif
}

// ---------------------------------------------------------------------------
// Kernel 0: pack/transpose weights to bf16, pack [q|c] rows to bf16
// ---------------------------------------------------------------------------
__global__ void pcr_pack_kernel(const float* __restrict__ query,
                                const float* __restrict__ condition,
                                const float* __restrict__ sw1,
                                const float* __restrict__ pw1,
                                const float* __restrict__ pw2,
                                __bf16* __restrict__ WtQC,   // [512][1024]
                                __bf16* __restrict__ WtS,    // [512][544]
                                __bf16* __restrict__ P1T,    // [512][544]
                                __bf16* __restrict__ P2T,    // [512][512]
                                __bf16* __restrict__ QCcat)  // [2048][1024]
{
    const size_t N0 = (size_t)EDIM * KQC;
    const size_t N1 = (size_t)EDIM * KEXT;
    const size_t N2 = (size_t)EDIM * KEXT;
    const size_t N3 = (size_t)EDIM * EDIM;
    const size_t N4 = (size_t)BATCH * KQC;
    size_t idx = (size_t)blockIdx.x * blockDim.x + threadIdx.x;
    if (idx < N0) {
        size_t n = idx / KQC, k = idx % KQC;
        WtQC[idx] = (__bf16)sw1[k * EDIM + n];               // sw1 rows 0..1023 (q|c)
    } else if ((idx -= N0) < N1) {
        size_t n = idx / KEXT, k = idx % KEXT;
        WtS[idx] = (k < 528) ? (__bf16)sw1[(1024 + k) * EDIM + n] : (__bf16)0.0f;
    } else if ((idx -= N1) < N2) {
        size_t n = idx / KEXT, k = idx % KEXT;
        P1T[idx] = (k < 528) ? (__bf16)pw1[k * EDIM + n] : (__bf16)0.0f;
    } else if ((idx -= N2) < N3) {
        size_t n = idx / EDIM, k = idx % EDIM;
        P2T[idx] = (__bf16)pw2[k * EDIM + n];
    } else if ((idx -= N3) < N4) {
        size_t row = idx / KQC, col = idx % KQC;
        float v = (col < EDIM) ? query[row * EDIM + col]
                               : condition[row * EDIM + (col - EDIM)];
        QCcat[idx] = (__bf16)v;
    }
}

// ---------------------------------------------------------------------------
// Kernel 1: qc[b,n] = (q|c)[b] @ WtQC^T + sb1   (64 batch rows per block)
// ---------------------------------------------------------------------------
__global__ __launch_bounds__(512)
void pcr_qc_kernel(const __bf16* __restrict__ QCcat,
                   const __bf16* __restrict__ WtQC,
                   const float* __restrict__ sb1,
                   float* __restrict__ qc) {
    __shared__ __align__(16) __bf16 Bs[2 * EDIM * 32];
    const int b0 = blockIdx.x * 64;
    v8f acc[8] = {};
    gemm64x512<KQC, KQC / 32>(QCcat + (size_t)b0 * KQC, WtQC, Bs, acc);

    const int lane = threadIdx.x & 31, wave = threadIdx.x >> 5;
    const int mt = wave & 3, nb = (wave >> 2) * 128;
    const int lr = lane & 15, kh = (lane >> 4) * 8;
#pragma unroll
    for (int nt = 0; nt < 8; ++nt) {
        const int n = nb + nt * 16 + lr;
        const float bias = sb1[n];
#pragma unroll
        for (int r = 0; r < 8; ++r) {
            const int m = mt * 16 + kh + r;
            qc[(size_t)(b0 + m) * EDIM + n] = acc[nt][r] + bias;
        }
    }
}

// ---------------------------------------------------------------------------
// Kernel 2: per-batch scorer GEMM + GELU + logit dot + softmax + mix
// ---------------------------------------------------------------------------
__global__ __launch_bounds__(512)
void pcr_main_kernel(const float* __restrict__ neighbors,
                     const float* __restrict__ stats,
                     const float* __restrict__ sw2,
                     const float* __restrict__ sb2p,
                     const float* __restrict__ alphap,
                     const __bf16* __restrict__ WtS,
                     const float* __restrict__ qc,
                     __bf16* __restrict__ mixed,
                     float* __restrict__ attn_out) {
    __shared__ __align__(16) __bf16 A[64 * KEXT];       // ~68 KB
    __shared__ __align__(16) __bf16 Bs[2 * EDIM * 32];  // 64 KB double-buffered
    __shared__ float logits[64];
    __shared__ float attn_sh[64];
    __shared__ float red[2];

    const int b = blockIdx.x;
    const int tid = threadIdx.x;

    // Stage A_ext = [neighbors | stats | 0-pad] as bf16 (8 threads per row)
    {
        const int row = tid >> 3;
        const int c0 = (tid & 7) * (KEXT / 8);   // 68 cols per thread
        const float* nrow = neighbors + ((size_t)b * KNBR + row) * EDIM;
        const float* srow = stats + ((size_t)b * KNBR + row) * SDIM;
        for (int j = 0; j < KEXT / 8; ++j) {
            const int col = c0 + j;
            float v = (col < EDIM) ? nrow[col]
                    : (col < EDIM + SDIM) ? srow[col - EDIM] : 0.0f;
            A[row * KEXT + col] = (__bf16)v;
        }
    }
    if (tid < 64) logits[tid] = 0.0f;

    v8f acc[8] = {};
    gemm64x512<KEXT, KEXT / 32>(A, WtS, Bs, acc);   // barriers inside

    const int lane = tid & 31, wave = tid >> 5;
    const int mt = wave & 3, nb = (wave >> 2) * 128;
    const int lr = lane & 15, kh = (lane >> 4) * 8;

    float lpart[8] = {};
#pragma unroll
    for (int nt = 0; nt < 8; ++nt) {
        const int n = nb + nt * 16 + lr;
        const float qn = qc[(size_t)b * EDIM + n];
        const float w2 = sw2[n];
#pragma unroll
        for (int r = 0; r < 8; ++r)
            lpart[r] += gelu_exact(acc[nt][r] + qn) * w2;
    }
#pragma unroll
    for (int r = 0; r < 8; ++r)
        atomicAdd(&logits[mt * 16 + kh + r], lpart[r]);
    __syncthreads();

    if (tid < 64)
        logits[tid] += sb2p[0] + alphap[0] * stats[((size_t)b * KNBR + tid) * SDIM + (SDIM - 1)];
    __syncthreads();

    if (tid == 0) {
        float mx = logits[0];
        for (int m = 1; m < 64; ++m) mx = fmaxf(mx, logits[m]);
        float s = 0.0f;
        for (int m = 0; m < 64; ++m) s += __expf(logits[m] - mx);
        red[0] = mx; red[1] = s;
    }
    __syncthreads();

    if (tid < 64) {
        const float a = __expf(logits[tid] - red[0]) / red[1];
        attn_sh[tid] = a;
        attn_out[(size_t)b * KNBR + tid] = a;
    }
    __syncthreads();

    for (int d = tid; d < KEXT; d += 512) {
        float s = 0.0f;
        for (int m = 0; m < 64; ++m)
            s += attn_sh[m] * (float)A[m * KEXT + d];
        mixed[(size_t)b * KEXT + d] = (__bf16)s;
    }
}

// ---------------------------------------------------------------------------
// Kernel 3: projection MLP: out = gelu(mixed @ pw1 + pb1) @ pw2 + pb2
// ---------------------------------------------------------------------------
__global__ __launch_bounds__(512)
void pcr_proj_kernel(const __bf16* __restrict__ mixed,
                     const __bf16* __restrict__ P1T,
                     const __bf16* __restrict__ P2T,
                     const float* __restrict__ pb1,
                     const float* __restrict__ pb2,
                     float* __restrict__ out) {
    __shared__ __align__(16) __bf16 Hs[64 * EDIM];      // 64 KB
    __shared__ __align__(16) __bf16 Bs[2 * EDIM * 32];  // 64 KB
    const int b0 = blockIdx.x * 64;

    const int lane = threadIdx.x & 31, wave = threadIdx.x >> 5;
    const int mt = wave & 3, nb = (wave >> 2) * 128;
    const int lr = lane & 15, kh = (lane >> 4) * 8;

    v8f acc[8] = {};
    gemm64x512<KEXT, KEXT / 32>(mixed + (size_t)b0 * KEXT, P1T, Bs, acc);
#pragma unroll
    for (int nt = 0; nt < 8; ++nt) {
        const int n = nb + nt * 16 + lr;
        const float bias = pb1[n];
#pragma unroll
        for (int r = 0; r < 8; ++r) {
            const int m = mt * 16 + kh + r;
            Hs[m * EDIM + n] = (__bf16)gelu_exact(acc[nt][r] + bias);
        }
    }
    v8f acc2[8] = {};
    gemm64x512<EDIM, EDIM / 32>(Hs, P2T, Bs, acc2);  // leading barrier publishes Hs
#pragma unroll
    for (int nt = 0; nt < 8; ++nt) {
        const int n = nb + nt * 16 + lr;
        const float bias = pb2[n];
#pragma unroll
        for (int r = 0; r < 8; ++r) {
            const int m = mt * 16 + kh + r;
            out[(size_t)(b0 + m) * EDIM + n] = acc2[nt][r] + bias;
        }
    }
}

// ---------------------------------------------------------------------------
// Launch
// ---------------------------------------------------------------------------
extern "C" void kernel_launch(void* const* d_in, const int* in_sizes, int n_in,
                              void* d_out, int out_size, void* d_ws, size_t ws_size,
                              hipStream_t stream) {
    const float* query     = (const float*)d_in[0];
    const float* condition = (const float*)d_in[1];
    const float* neighbors = (const float*)d_in[2];
    const float* stats     = (const float*)d_in[3];
    const float* sw1       = (const float*)d_in[4];
    const float* sb1       = (const float*)d_in[5];
    const float* sw2       = (const float*)d_in[6];
    const float* sb2       = (const float*)d_in[7];
    const float* alpha     = (const float*)d_in[8];
    const float* pw1       = (const float*)d_in[9];
    const float* pb1       = (const float*)d_in[10];
    const float* pw2       = (const float*)d_in[11];
    const float* pb2       = (const float*)d_in[12];

    char* ws = (char*)d_ws;
    size_t o = 0;
    __bf16* WtQC  = (__bf16*)(ws + o); o += (size_t)EDIM * KQC  * 2;
    __bf16* WtS   = (__bf16*)(ws + o); o += (size_t)EDIM * KEXT * 2;
    __bf16* P1T   = (__bf16*)(ws + o); o += (size_t)EDIM * KEXT * 2;
    __bf16* P2T   = (__bf16*)(ws + o); o += (size_t)EDIM * EDIM * 2;
    __bf16* QCcat = (__bf16*)(ws + o); o += (size_t)BATCH * KQC * 2;
    float*  qcbuf = (float*)(ws + o);  o += (size_t)BATCH * EDIM * 4;
    __bf16* mixed = (__bf16*)(ws + o); o += (size_t)BATCH * KEXT * 2;

    float* outp  = (float*)d_out;
    float* attnp = outp + (size_t)BATCH * EDIM;

    const size_t pack_elems = (size_t)EDIM * KQC + 2 * (size_t)EDIM * KEXT +
                              (size_t)EDIM * EDIM + (size_t)BATCH * KQC;
    const int pack_blocks = (int)((pack_elems + 255) / 256);

    pcr_pack_kernel<<<pack_blocks, 256, 0, stream>>>(query, condition, sw1, pw1, pw2,
                                                     WtQC, WtS, P1T, P2T, QCcat);
    pcr_qc_kernel<<<BATCH / 64, 512, 0, stream>>>(QCcat, WtQC, sb1, qcbuf);
    pcr_main_kernel<<<BATCH, 512, 0, stream>>>(neighbors, stats, sw2, sb2, alpha,
                                               WtS, qcbuf, mixed, attnp);
    pcr_proj_kernel<<<BATCH / 64, 512, 0, stream>>>(mixed, P1T, P2T, pb1, pb2, outp);
}
